// FP8Linear_36507222016222
// MI455X (gfx1250) — 8-bit matrix operand order repaired and checked by reading, compiled, not run on hardware
//
#include <hip/hip_runtime.h>
#include <hip/hip_bf16.h>


typedef __attribute__((ext_vector_type(16))) int   v16i;
typedef __attribute__((ext_vector_type(8)))  float v8f;

#define K_DIM 1024
#define N_DIM 1024

__global__ void fp8lin_init_kernel(unsigned int* amax_bits) {
    if (threadIdx.x == 0) *amax_bits = 0u;
}

__global__ __launch_bounds__(256)
void fp8lin_amax_kernel(const float* __restrict__ x, unsigned int* amax_bits, int n4) {
    const float4* x4 = (const float4*)x;
    float m = 0.0f;
    for (int i = blockIdx.x * blockDim.x + threadIdx.x; i < n4;
         i += gridDim.x * blockDim.x) {
        float4 v = x4[i];
        m = fmaxf(m, fmaxf(fmaxf(fabsf(v.x), fabsf(v.y)),
                           fmaxf(fabsf(v.z), fabsf(v.w))));
    }
    #pragma unroll
    for (int off = 16; off > 0; off >>= 1)
        m = fmaxf(m, __shfl_xor(m, off, 32));
    if ((threadIdx.x & 31) == 0)
        atomicMax(amax_bits, __float_as_uint(m));
}

__global__ __launch_bounds__(256)
void fp8lin_quant_kernel(const float* __restrict__ x,
                         unsigned char* __restrict__ x8,
                         const unsigned int* __restrict__ amax_bits) {
    const int i = (blockIdx.x * 256 + threadIdx.x) * 8;
    const float amax = fmaxf(__uint_as_float(*amax_bits), 1e-12f);
    const float s = 448.0f / amax;

    float4 a = *(const float4*)(x + i);
    float4 b = *(const float4*)(x + i + 4);
    float q0 = fminf(fmaxf(a.x * s, -448.0f), 448.0f);
    float q1 = fminf(fmaxf(a.y * s, -448.0f), 448.0f);
    float q2 = fminf(fmaxf(a.z * s, -448.0f), 448.0f);
    float q3 = fminf(fmaxf(a.w * s, -448.0f), 448.0f);
    float q4 = fminf(fmaxf(b.x * s, -448.0f), 448.0f);
    float q5 = fminf(fmaxf(b.y * s, -448.0f), 448.0f);
    float q6 = fminf(fmaxf(b.z * s, -448.0f), 448.0f);
    float q7 = fminf(fmaxf(b.w * s, -448.0f), 448.0f);

    int w0 = __builtin_amdgcn_cvt_pk_fp8_f32(q0, q1, 0,  false);
    w0     = __builtin_amdgcn_cvt_pk_fp8_f32(q2, q3, w0, true);
    int w1 = __builtin_amdgcn_cvt_pk_fp8_f32(q4, q5, 0,  false);
    w1     = __builtin_amdgcn_cvt_pk_fp8_f32(q6, q7, w1, true);

    uint2 out;
    out.x = (unsigned int)w0;
    out.y = (unsigned int)w1;
    *(uint2*)(x8 + i) = out;
}

__global__ __launch_bounds__(256)
void fp8lin_gemm_kernel(const unsigned char* __restrict__ A8,
                        const unsigned char* __restrict__ W8,
                        const unsigned int*  __restrict__ amax_bits,
                        const float*         __restrict__ w_scale,
                        __hip_bfloat16*      __restrict__ out) {
    const int lane = threadIdx.x & 31;
    const int wave = threadIdx.x >> 5;
    const int l15  = lane & 15;
    const int h    = lane >> 4;
    const int wm   = wave & 1;
    const int wn   = wave >> 1;
    const int mBase = blockIdx.y * 32 + wm * 16;
    const int nBase = blockIdx.x * 256 + wn * 64;

    union Areg { v16i v; uint2 d[8]; };
    union Breg { v16i v; uint4 q[4]; };

    v8f acc[4] = {};
    Areg a[2];
    Breg b[4];

    const unsigned char* aPtr = A8 + (size_t)(mBase + l15) * K_DIM + h * 8;
    const unsigned char* bRow = W8 + (size_t)(nBase + l15) * K_DIM + h * 16;

    auto loadA = [&](Areg& dst, int kk) {
        #pragma unroll
        for (int t = 0; t < 8; ++t) {
            const int off = ((t >> 1) << 5) + ((t & 1) << 3) + (h << 3);
            dst.d[t] = *(const uint2*)(aPtr + kk + off);
        }
    };
    auto loadB = [&](Breg& dst, int kk, int j) {
        const unsigned char* p = bRow + (size_t)j * (16 * K_DIM) + kk;
        #pragma unroll
        for (int q = 0; q < 4; ++q)
            dst.q[q] = *(const uint4*)(p + (q << 5));
    };

    loadA(a[0], 0);
    loadB(b[0], 0, 0);
    loadB(b[1], 0, 1);
    loadB(b[2], 0, 2);

    #pragma unroll
    for (int step = 0; step < 32; ++step) {
        const int j = step & 3;
        if (j == 0 && step < 28) {
            const int nk = (step >> 2) + 1;
            loadA(a[nk & 1], nk << 7);
        }
        const int pf = step + 3;
        if (pf < 32)
            loadB(b[pf & 3], (pf >> 2) << 7, pf & 3);

        acc[j] = __builtin_amdgcn_wmma_f32_16x16x128_fp8_fp8(
            a[(step >> 2) & 1].v, b[j].v, (short)0, acc[j], false, false);
    }

    const float amax = fmaxf(__uint_as_float(*amax_bits), 1e-12f);
    const float cscale = (1.0f / (448.0f / amax)) * (*w_scale);

    unsigned short* outs = (unsigned short*)out;
    #pragma unroll
    for (int j = 0; j < 4; ++j) {
        const int n = nBase + j * 16 + l15;
        #pragma unroll
        for (int r = 0; r < 8; ++r) {
            const int m = mBase + r + 8 * h;
            const unsigned short bits =
                __builtin_bit_cast(unsigned short, __float2bfloat16(acc[j][r] * cscale));
            __builtin_nontemporal_store(bits, outs + (size_t)m * N_DIM + n);
        }
    }
}

extern "C" void kernel_launch(void* const* d_in, const int* in_sizes, int n_in,
                              void* d_out, int out_size, void* d_ws, size_t ws_size,
                              hipStream_t stream) {
    const float*         x      = (const float*)d_in[0];
    const unsigned char* w8     = (const unsigned char*)d_in[1];
    const float*         wscale = (const float*)d_in[2];
    __hip_bfloat16*      out    = (__hip_bfloat16*)d_out;

    const int nElem = in_sizes[0];
    const int S     = nElem / K_DIM;

    unsigned int*  amax_bits = (unsigned int*)d_ws;
    unsigned char* x8        = (unsigned char*)d_ws + 256;

    fp8lin_init_kernel<<<1, 32, 0, stream>>>(amax_bits);
    fp8lin_amax_kernel<<<2048, 256, 0, stream>>>(x, amax_bits, nElem / 4);
    fp8lin_quant_kernel<<<nElem / 2048, 256, 0, stream>>>(x, x8, amax_bits);

    dim3 grid(N_DIM / 256, S / 32);
    fp8lin_gemm_kernel<<<grid, 256, 0, stream>>>(x8, w8, amax_bits, wscale, out);
}
